// Crf_56882546868901
// MI455X (gfx1250) — matrix operand order repaired and checked by reading, not compiled, not run on hardware
//
#include <hip/hip_runtime.h>

// CRF forward (negative log-likelihood) for B=512, S=512, L=128 on gfx1250.
//
// alpha[b,i] = log(U[b,i]) + c[b];  per step:
//     W = (U @ E) * exp(x_t),  E = exp(trans) (constant 128x128)
// renormalize (U /= rowmax, c += log rowmax) only every 4th step — the
// invariant makes skipped normalizations free; growth <= ~(5e4)^4 stays
// far below f32/bf16 range.
//
// GEMM maps to v_wmma_f32_16x16x32_bf16: 4 K-slices x 8 N-tiles = 32 WMMAs
// per step per 16-batch wave. All 32 E B-fragments (256 VGPRs) are kept
// resident in the extended wave32 VGPR file (up to 1024 VGPRs/wave) for the
// entire scan; only the small U tile round-trips through LDS.

#define LDIM 128
#define SDIM 512
#define BDIM 512

typedef __bf16 v16bf __attribute__((ext_vector_type(16)));
typedef __bf16 v8bf  __attribute__((ext_vector_type(8)));
typedef float  v8f   __attribute__((ext_vector_type(8)));

// Reduce across the 16 lanes of a half-wave (xor masks 8,4,2,1 stay inside
// the 16-lane group). ds_swizzle offset: [14:10]=xor, [9:5]=or, [4:0]=and.
__device__ __forceinline__ float red16_max(float v) {
    v = fmaxf(v, __int_as_float(__builtin_amdgcn_ds_swizzle(__float_as_int(v), 0x201F)));
    v = fmaxf(v, __int_as_float(__builtin_amdgcn_ds_swizzle(__float_as_int(v), 0x101F)));
    v = fmaxf(v, __int_as_float(__builtin_amdgcn_ds_swizzle(__float_as_int(v), 0x081F)));
    v = fmaxf(v, __int_as_float(__builtin_amdgcn_ds_swizzle(__float_as_int(v), 0x041F)));
    return v;
}
__device__ __forceinline__ float red16_sum(float v) {
    v += __int_as_float(__builtin_amdgcn_ds_swizzle(__float_as_int(v), 0x201F));
    v += __int_as_float(__builtin_amdgcn_ds_swizzle(__float_as_int(v), 0x101F));
    v += __int_as_float(__builtin_amdgcn_ds_swizzle(__float_as_int(v), 0x081F));
    v += __int_as_float(__builtin_amdgcn_ds_swizzle(__float_as_int(v), 0x041F));
    return v;
}

// ---------------------------------------------------------------------------
// Kernel 1: label_score[b] = point_score + trans_score  (memory-bound gathers)
// ---------------------------------------------------------------------------
__global__ void label_score_kernel(const float* __restrict__ x,
                                   const int*   __restrict__ label,
                                   const int*   __restrict__ length,
                                   const float* __restrict__ trans,
                                   float*       __restrict__ ws) {
    const int b = blockIdx.x;
    int len = length[b];
    len = len < 1 ? 1 : (len > SDIM ? SDIM : len);
    float acc = 0.0f;
    for (int t = threadIdx.x; t < len; t += blockDim.x) {
        const int lab = label[b * SDIM + t];
        acc += x[((size_t)b * SDIM + t) * LDIM + lab];
        if (t + 1 < len) {
            const int lab2 = label[b * SDIM + t + 1];
            acc += trans[lab * LDIM + lab2];
        }
    }
    __shared__ float red[256];
    red[threadIdx.x] = acc;
    __syncthreads();
    for (int ofs = 128; ofs > 0; ofs >>= 1) {
        if ((int)threadIdx.x < ofs) red[threadIdx.x] += red[threadIdx.x + ofs];
        __syncthreads();
    }
    if (threadIdx.x == 0) ws[b] = red[0];
}

// ---------------------------------------------------------------------------
// Kernel 2: WMMA forward scan. 1 wave (32 threads) per block, 16 batches each.
// ---------------------------------------------------------------------------
__global__ __launch_bounds__(32)
void crf_forward_kernel(const float* __restrict__ x,
                        const int*   __restrict__ length,
                        const float* __restrict__ trans,
                        const float* __restrict__ lscore,
                        float*       __restrict__ out) {
    // Current U (bf16), row-major [batch][state], padded pitch for bank spread.
    __shared__ __align__(16) __bf16 Ubuf[16][136];   // 4.25 KB (only LDS use)

    const int lane = threadIdx.x;        // 0..31 (wave32)
    const int lm   = lane & 15;          // column / position within half-wave
    const int h    = lane >> 4;          // half-wave select
    const int b0   = blockIdx.x * 16;    // first batch of this tile

    // ---- E = exp(trans) as register-resident bf16 B fragments ----
    // Fragment (kk, n): K = 32*kk..32*kk+31, N = 16*n..16*n+15.
    // B fragments use contiguous K (same order as A): lanes 0-15 hold K = Kbase+0..15, lanes 16-31
    // hold K = Kbase+16..31, column = lane%16, 2 K values per dword.
    // Element e of the per-lane v16bf <-> Klocal = e + (h ? 16 : 0).
    v16bf Ereg[4][8];
#pragma unroll
    for (int kk = 0; kk < 4; ++kk)
#pragma unroll
        for (int n = 0; n < 8; ++n) {
            v16bf f;
#pragma unroll
            for (int e = 0; e < 16; ++e) {
                const int i = kk * 32 + (h ? 16 : 0) + e;   // K (row of E)
                f[e] = (__bf16)__expf(trans[i * LDIM + n * 16 + lm]);
            }
            Ereg[kk][n] = f;
        }

    // Per-lane batch rows: C/D tile layout puts (batch k+8h, col 16n+lm) in
    // acc[n] element k on lane (lm + 16h).
    const float* rp[8];
    int   lenr[8];
    float c[8];
#pragma unroll
    for (int k = 0; k < 8; ++k) {
        const int b = b0 + k + 8 * h;
        int L_ = length[b];
        lenr[k] = L_ < 1 ? 1 : (L_ > SDIM ? SDIM : L_);
        rp[k]   = x + (size_t)b * SDIM * LDIM + lm;
    }

    // ---- t = 0: alpha0 = x[:,0,:]  ->  U = exp(alpha0 - m), c = m ----
    {
        float w[8][8];
#pragma unroll
        for (int n = 0; n < 8; ++n)
#pragma unroll
            for (int k = 0; k < 8; ++k) w[n][k] = rp[k][n * 16];
#pragma unroll
        for (int k = 0; k < 8; ++k) {
            float m = w[0][k];
#pragma unroll
            for (int n = 1; n < 8; ++n) m = fmaxf(m, w[n][k]);
            m    = red16_max(m);
            c[k] = m;
#pragma unroll
            for (int n = 0; n < 8; ++n)
                Ubuf[k + 8 * h][n * 16 + lm] = (__bf16)__expf(w[n][k] - m);
        }
    }
    __syncthreads();

    // ---- sequential scan: t = 1 .. S-1 ----
    for (int t = 1; t < SDIM; ++t) {
        // Prefetch x_t (coalesced: each (n,k) load is two contiguous 64B runs).
        float xr[8][8];
        const int tofs = t * LDIM;
#pragma unroll
        for (int n = 0; n < 8; ++n)
#pragma unroll
            for (int k = 0; k < 8; ++k) xr[n][k] = rp[k][tofs + n * 16];

        // GEMM: W(16x128) = U(16x128) @ E(128x128), f32 accumulate.
        v8f acc[8];
#pragma unroll
        for (int n = 0; n < 8; ++n) acc[n] = (v8f){0, 0, 0, 0, 0, 0, 0, 0};
#pragma unroll
        for (int kk = 0; kk < 4; ++kk) {
            // A fragment (16x32 bf16): row = lane%16 (batch), per-lane K run
            // of 16 contiguous states (K = 16h + e), matching the B fragments.
            const v8bf alo = *(const v8bf*)&Ubuf[lm][kk * 32 + 16 * h];
            const v8bf ahi = *(const v8bf*)&Ubuf[lm][kk * 32 + 16 * h + 8];
            const v16bf A  = __builtin_shufflevector(alo, ahi, 0, 1, 2, 3, 4, 5, 6, 7,
                                                     8, 9, 10, 11, 12, 13, 14, 15);
#pragma unroll
            for (int n = 0; n < 8; ++n)
                acc[n] = __builtin_amdgcn_wmma_f32_16x16x32_bf16(
                    false, A, false, Ereg[kk][n], (short)0, acc[n], false, false);
        }

        // W *= exp(x_t)
#pragma unroll
        for (int n = 0; n < 8; ++n)
#pragma unroll
            for (int k = 0; k < 8; ++k) acc[n][k] *= __expf(xr[n][k]);

        if ((t & 3) == 0) {
            // Renormalize + write back (rows freeze once t >= length[b]).
#pragma unroll
            for (int k = 0; k < 8; ++k) {
                float m = acc[0][k];
#pragma unroll
                for (int n = 1; n < 8; ++n) m = fmaxf(m, acc[n][k]);
                m = red16_max(m);                   // per-batch row max
                const float r = __builtin_amdgcn_rcpf(m);
                if (t < lenr[k]) {
                    c[k] += __logf(m);
#pragma unroll
                    for (int n = 0; n < 8; ++n)
                        Ubuf[k + 8 * h][n * 16 + lm] = (__bf16)(acc[n][k] * r);
                }
            }
        } else {
            // No normalization: alpha = log(U) + c invariant makes this exact.
#pragma unroll
            for (int k = 0; k < 8; ++k) {
                if (t < lenr[k]) {
#pragma unroll
                    for (int n = 0; n < 8; ++n)
                        Ubuf[k + 8 * h][n * 16 + lm] = (__bf16)acc[n][k];
                }
            }
        }
        __syncthreads();   // S_NOP for single-wave WG; orders LDS store->load
    }

    // ---- finalize: z = c + log(sum_j U);  out = z - label_score ----
    float zk[8];
#pragma unroll
    for (int k = 0; k < 8; ++k) {
        float s = 0.0f;
#pragma unroll
        for (int n = 0; n < 8; ++n) s += (float)Ubuf[k + 8 * h][n * 16 + lm];
        s     = red16_sum(s);
        zk[k] = c[k] + __logf(s);
    }
#pragma unroll
    for (int k = 0; k < 8; ++k) {
        const int b = b0 + k + 8 * h;
        if (lm == k) out[b] = zk[k] - lscore[b];
    }
}

// ---------------------------------------------------------------------------
extern "C" void kernel_launch(void* const* d_in, const int* in_sizes, int n_in,
                              void* d_out, int out_size, void* d_ws, size_t ws_size,
                              hipStream_t stream) {
    const float* input  = (const float*)d_in[0];   // (B,S,L) f32
    const int*   label  = (const int*)d_in[1];     // (B,S)   i32
    const int*   length = (const int*)d_in[2];     // (B,)    i32
    const float* trans  = (const float*)d_in[3];   // (L,L)   f32
    float*       out    = (float*)d_out;           // (B,1)   f32
    float*       lsc    = (float*)d_ws;            // B floats of scratch

    label_score_kernel<<<BDIM, 256, 0, stream>>>(input, label, length, trans, lsc);
    crf_forward_kernel<<<BDIM / 16, 32, 0, stream>>>(input, length, trans, lsc, out);
}
